// SchEmb_79121887527182
// MI455X (gfx1250) — compile-verified, matrix operand order checked by reading, not run on hardware
//
#include <hip/hip_runtime.h>

#define NN 50000     // N_NODES
#define NE 200000    // N_EDGES
#define NG 1024      // N_GRAPH
#define HC 128       // hidden channels H
#define VC 80        // vert-emb channels (H - POS)
#define NF 256       // num_filters
#define LAYERS 5
#define NFF 512

typedef __attribute__((ext_vector_type(16))) __bf16 v16bf;
typedef __attribute__((ext_vector_type(8)))  float  v8f;
typedef __attribute__((ext_vector_type(4)))  int    v4i_t;

union FragBF {
    v16bf bf;
    uint4 q[2];
};

union PackBF {
    __bf16 h[2];
    unsigned int u;
};

__device__ __forceinline__ float ssp_f(float v) {
    // shifted softplus: softplus(v) - ln2, numerically stable
    return fmaxf(v, 0.f) + log1pf(expf(-fabsf(v))) - 0.69314718055994531f;
}

// ---------------------------------------------------------------------------
// bf16 WMMA GEMM:  out = act(A[M,K] @ B[K,Nc] + bias) (+ resid)
//   A : bf16 row-major (rows 16B-aligned since K % 8 == 0)
//   Bp: fragment-packed weights, tile t = n_tile*(K/32)+k_tile, 256 uints/tile,
//       within tile: [lane][vgpr] so a lane reads 8 consecutive uints.
// One wave -> 16x32 of C (two 16x16 tiles sharing the A frag).
// Block = 8 waves -> 128 x 32 region. B staged through LDS (shared by all
// waves), via CDNA5 async global->LDS copies (ASYNCcnt) when available.
// ACT: 0=none 1=ssp 2=relu+resid 3=tanh+resid 4=gelu(exact)
// ---------------------------------------------------------------------------
template <int K, int ACT>
__global__ __launch_bounds__(256)
void wmma_gemm_kernel(const __bf16* __restrict__ A, const unsigned int* __restrict__ Bp,
                      const float* __restrict__ bias, const float* __restrict__ resid,
                      float* __restrict__ outF, __bf16* __restrict__ outB,
                      int M, int Nc)
{
    constexpr int KT = K / 32;                 // k-tiles
    __shared__ unsigned int Bs[2 * KT * 256];  // two N-subtiles worth of packed B

    // ---- cooperative stage of packed B (contiguous: n_tile and n_tile+1) ----
    {
        const uint4* src = (const uint4*)(Bp + (size_t)(blockIdx.y * 2) * KT * 256);
        uint4* dstp = (uint4*)Bs;
        const int total4 = (2 * KT * 256) / 4;
#if defined(__AMDGCN__) && __has_builtin(__builtin_amdgcn_global_load_async_to_lds_b128)
        for (int i = threadIdx.x; i < total4; i += 256) {
            __builtin_amdgcn_global_load_async_to_lds_b128(
                (__attribute__((address_space(1))) v4i_t*)(src + i),
                (__attribute__((address_space(3))) v4i_t*)(dstp + i),
                /*offset=*/0, /*cpol=*/0);
        }
#if __has_builtin(__builtin_amdgcn_s_wait_asynccnt)
        __builtin_amdgcn_s_wait_asynccnt(0);
#else
        asm volatile("s_wait_asynccnt 0x0" ::: "memory");
#endif
#else
        for (int i = threadIdx.x; i < total4; i += 256) dstp[i] = src[i];
#endif
    }
    __syncthreads();

    const int lane  = threadIdx.x & 31;
    const int wid   = threadIdx.x >> 5;
    const int tileM = blockIdx.x * 128 + wid * 16;
    const int half  = lane >> 4;
    const int l16   = lane & 15;

    // branchless row clamp: OOB rows read row M-1; their C rows are never stored
    int m = tileM + l16;
    if (m >= M) m = M - 1;
    const __bf16* arow = A + (size_t)m * K;

    v8f c0 = {0.f, 0.f, 0.f, 0.f, 0.f, 0.f, 0.f, 0.f};
    v8f c1 = {0.f, 0.f, 0.f, 0.f, 0.f, 0.f, 0.f, 0.f};

#pragma unroll
    for (int kt = 0; kt < KT; ++kt) {
        FragBF a, b0, b1;
        // A frag: two contiguous 16B chunks (ISA 16-bit A 16x32 layout)
        const __bf16* ap = arow + kt * 32 + half * 8;
        a.q[0] = *(const uint4*)(ap);
        a.q[1] = *(const uint4*)(ap + 16);
        // B frags from LDS: 8 consecutive uints per lane per subtile
        const unsigned int* bs0 = Bs + kt * 256 + lane * 8;
        const unsigned int* bs1 = bs0 + KT * 256;
        b0.q[0] = *(const uint4*)(bs0);
        b0.q[1] = *(const uint4*)(bs0 + 4);
        b1.q[0] = *(const uint4*)(bs1);
        b1.q[1] = *(const uint4*)(bs1 + 4);

        c0 = __builtin_amdgcn_wmma_f32_16x16x32_bf16(false, a.bf, false, b0.bf,
                                                     (short)0, c0, false, false);
        c1 = __builtin_amdgcn_wmma_f32_16x16x32_bf16(false, a.bf, false, b1.bf,
                                                     (short)0, c1, false, false);
    }

    // ---- epilogue: C/D f32 layout: VGPR v -> row half*8+v, col lane&15 ----
    const int n0 = blockIdx.y * 32 + l16;
    const int n1 = n0 + 16;
    const float bn0 = bias ? bias[n0] : 0.f;
    const float bn1 = bias ? bias[n1] : 0.f;
#pragma unroll
    for (int v = 0; v < 8; ++v) {
        const int row = tileM + half * 8 + v;
        if (row >= M) continue;
        const size_t i0 = (size_t)row * Nc + n0;
        const size_t i1 = (size_t)row * Nc + n1;
        float v0 = c0[v] + bn0;
        float v1 = c1[v] + bn1;
        if (ACT == 1) { v0 = ssp_f(v0); v1 = ssp_f(v1); }
        if (ACT == 2) { v0 = fmaxf(v0, 0.f) + resid[i0]; v1 = fmaxf(v1, 0.f) + resid[i1]; }
        if (ACT == 3) { v0 = tanhf(v0) + resid[i0]; v1 = tanhf(v1) + resid[i1]; }
        if (ACT == 4) {
            v0 = 0.5f * v0 * (1.f + erff(v0 * 0.70710678118654752f));
            v1 = 0.5f * v1 * (1.f + erff(v1 * 0.70710678118654752f));
        }
        if (outF) { outF[i0] = v0; outF[i1] = v1; }
        if (outB) { outB[i0] = (__bf16)v0; outB[i1] = (__bf16)v1; }
    }
}

// ---------------------------------------------------------------------------
// Weight pack: fp32 W[out=Nc, in=K] -> fragment-ordered packed bf16 pairs.
// ---------------------------------------------------------------------------
__global__ void wpack_kernel(const float* __restrict__ w, unsigned int* __restrict__ o,
                             int K, int Nc) {
    const int total = (K * Nc) >> 1;
    int i = blockIdx.x * blockDim.x + threadIdx.x;
    if (i >= total) return;
    const int ktiles = K >> 5;
    const int t = i >> 8;            // tile index
    const int r = i & 255;
    const int l = r >> 3;            // lane
    const int v = r & 7;             // vgpr slot
    const int k_tile = t % ktiles;
    const int n_tile = t / ktiles;
    const int half = l >> 4, l16 = l & 15;
    const int n = n_tile * 16 + l16;
    const int k = (k_tile << 5) + ((v >> 2) << 4) + (half << 3) + ((v & 3) << 1);
    PackBF pk;
    pk.h[0] = (__bf16)w[(size_t)n * K + k];
    pk.h[1] = (__bf16)w[(size_t)n * K + k + 1];
    o[i] = pk.u;
}

// ---------------------------------------------------------------------------
// Elementwise / irregular kernels
// ---------------------------------------------------------------------------
__global__ void cast4_kernel(const float4* __restrict__ in, uint2* __restrict__ out, int cnt4) {
    int i = blockIdx.x * blockDim.x + threadIdx.x;
    if (i >= cnt4) return;
    float4 f = in[i];
    PackBF a, b;
    a.h[0] = (__bf16)f.x; a.h[1] = (__bf16)f.y;
    b.h[0] = (__bf16)f.z; b.h[1] = (__bf16)f.w;
    out[i] = make_uint2(a.u, b.u);
}

__global__ void zero_kernel(float* __restrict__ p, int cnt) {
    int i = blockIdx.x * blockDim.x + threadIdx.x;
    if (i < cnt) p[i] = 0.f;
}

// h (f32 + bf16) = concat(vert_emb[x], pos @ pos_w.T)
__global__ void embed_kernel(const int* __restrict__ x, const float* __restrict__ pos,
                             const float* __restrict__ vert, const float* __restrict__ posw,
                             float* __restrict__ h, __bf16* __restrict__ hb) {
    int i = blockIdx.x * blockDim.x + threadIdx.x;
    if (i >= NN * HC) return;
    int nd = i >> 7, c = i & (HC - 1);
    float v;
    if (c < VC) {
        v = vert[(size_t)x[nd] * VC + c];
    } else {
        int cc = c - VC;
        const float* pr = pos + (size_t)nd * 3;
        const float* wr = posw + (size_t)cc * 3;
        v = pr[0] * wr[0] + pr[1] * wr[1] + pr[2] * wr[2];
    }
    h[i] = v;
    hb[i] = (__bf16)v;
}

__global__ void edgeproj_kernel(const float* __restrict__ eattr, const float* __restrict__ ew,
                                float* __restrict__ ea, __bf16* __restrict__ eab) {
    int i = blockIdx.x * blockDim.x + threadIdx.x;
    if (i >= NE * HC) return;
    int e = i >> 7, c = i & (HC - 1);
    const float* ar = eattr + (size_t)e * 4;
    const float* wr = ew + (size_t)c * 4;
    float v = ar[0] * wr[0] + ar[1] * wr[1] + ar[2] * wr[2] + ar[3] * wr[3];
    ea[i] = v;
    eab[i] = (__bf16)v;
}

// X[e, 0:128] = ea[e], X[e, 128:256] = h[src[e]] + h[dst[e]]   (bf16)
__global__ void econcat_kernel(const float* __restrict__ ea, const float* __restrict__ h,
                               const int* __restrict__ src, const int* __restrict__ dst,
                               __bf16* __restrict__ X) {
    int i = blockIdx.x * blockDim.x + threadIdx.x;
    if (i >= NE * 2 * HC) return;
    int e = i >> 8, c = i & (2 * HC - 1);
    float v;
    if (c < HC) v = ea[(size_t)e * HC + c];
    else {
        int cc = c - HC;
        v = h[(size_t)src[e] * HC + cc] + h[(size_t)dst[e] * HC + cc];
    }
    X[i] = (__bf16)v;
}

// agg[dst[e], c] += lin1h[src[e], c] * Wf[e, c]
__global__ void cfconv_scatter_kernel(const float* __restrict__ lin1h,
                                      const float* __restrict__ Wf,
                                      const int* __restrict__ src, const int* __restrict__ dst,
                                      float* __restrict__ agg) {
    int i = blockIdx.x * blockDim.x + threadIdx.x;
    if (i >= NE * NF) return;
    int e = i >> 8, c = i & (NF - 1);
    float v = lin1h[(size_t)src[e] * NF + c] * Wf[i];
    atomicAdd(&agg[(size_t)dst[e] * NF + c], v);
}

__global__ void pool_scatter_kernel(const float* __restrict__ h, const int* __restrict__ batch,
                                    float* __restrict__ ssum, float* __restrict__ cnt) {
    int i = blockIdx.x * blockDim.x + threadIdx.x;
    if (i >= NN * HC) return;
    int nd = i >> 7, c = i & (HC - 1);
    int g = batch[nd];
    atomicAdd(&ssum[(size_t)g * HC + c], h[i]);
    if (c == 0) atomicAdd(&cnt[g], 1.0f);
}

__global__ void pool_fin_kernel(const float* __restrict__ ssum, const float* __restrict__ cnt,
                                __bf16* __restrict__ gb) {
    int i = blockIdx.x * blockDim.x + threadIdx.x;
    if (i >= NG * HC) return;
    int g = i >> 7;
    gb[i] = (__bf16)(ssum[i] / fmaxf(cnt[g], 1.0f));
}

__global__ void head2_kernel(const float* __restrict__ h1, const float* __restrict__ w2,
                             const float* __restrict__ b2, float* __restrict__ out) {
    int g = blockIdx.x * blockDim.x + threadIdx.x;
    if (g >= NG) return;
    float s = 0.f;
    const float* r = h1 + (size_t)g * NFF;
    for (int j = 0; j < NFF; ++j) s += r[j] * w2[j];
    out[g] = s + b2[0];
}

// ---------------------------------------------------------------------------
// Host orchestration
// ---------------------------------------------------------------------------
extern "C" void kernel_launch(void* const* d_in, const int* in_sizes, int n_in,
                              void* d_out, int out_size, void* d_ws, size_t ws_size,
                              hipStream_t stream) {
    const int*   x     = (const int*)d_in[0];
    const int*   eidx  = (const int*)d_in[1];
    const int*   src   = eidx;
    const int*   dst   = eidx + NE;
    const float* eattr = (const float*)d_in[2];
    const int*   batch = (const int*)d_in[3];
    const float* pos   = (const float*)d_in[4];
    const float* vert  = (const float*)d_in[5];
    const float* posw  = (const float*)d_in[6];
    const float* edgew = (const float*)d_in[7];
    const float* fw1   = (const float*)d_in[8];
    const float* fb1   = (const float*)d_in[9];
    const float* fw2   = (const float*)d_in[10];
    const float* fb2   = (const float*)d_in[11];
    const float* l1w   = (const float*)d_in[12];
    const float* l2w   = (const float*)d_in[13];
    const float* l2b   = (const float*)d_in[14];
    const float* l3w   = (const float*)d_in[15];
    const float* l3b   = (const float*)d_in[16];
    const float* emw   = (const float*)d_in[17];
    const float* emb   = (const float*)d_in[18];
    const float* hw1   = (const float*)d_in[19];
    const float* hb1   = (const float*)d_in[20];
    const float* hw2   = (const float*)d_in[21];
    const float* hb2   = (const float*)d_in[22];
    float* y = (float*)d_out;

    // ---- carve workspace ----
    char* p = (char*)d_ws;
    auto take = [&](size_t bytes) -> char* {
        char* r = p;
        p += (bytes + 255) & ~(size_t)255;
        return r;
    };
    typedef unsigned int u32;
    u32* WBf1 = (u32*)take((size_t)LAYERS * HC * NF * 2);     // packed [K=128,N=256] x5
    u32* WBf2 = (u32*)take((size_t)LAYERS * NF * NF * 2);     // [256,256] x5
    u32* WBl1 = (u32*)take((size_t)LAYERS * HC * NF * 2);     // [128,256] x5
    u32* WBl2 = (u32*)take((size_t)LAYERS * NF * HC * 2);     // [256,128] x5
    u32* WBl3 = (u32*)take((size_t)LAYERS * HC * HC * 2);     // [128,128] x5
    u32* WBem = (u32*)take((size_t)LAYERS * 2 * HC * HC * 2); // [256,128] x5
    u32* WBh1 = (u32*)take((size_t)HC * NFF * 2);             // [128,512]

    float*  hbuf  = (float*)take((size_t)NN * HC * 4);
    float*  ea    = (float*)take((size_t)NE * HC * 4);
    __bf16* ea_b  = (__bf16*)take((size_t)NE * HC * 2);
    __bf16* t1_b  = (__bf16*)take((size_t)NE * NF * 2);
    float*  Wf    = (float*)take((size_t)NE * NF * 4);
    __bf16* h_b   = (__bf16*)take((size_t)NN * HC * 2);
    float*  lin1h = (float*)take((size_t)NN * NF * 4);
    float*  agg   = (float*)take((size_t)NN * NF * 4);
    __bf16* agg_b = (__bf16*)take((size_t)NN * NF * 2);
    __bf16* t2_b  = (__bf16*)take((size_t)NN * HC * 2);
    __bf16* Xb    = (__bf16*)take((size_t)NE * 2 * HC * 2);
    float*  ssum  = (float*)take((size_t)NG * HC * 4);
    float*  cnt   = (float*)take((size_t)NG * 4);
    __bf16* g_b   = (__bf16*)take((size_t)NG * HC * 2);
    float*  h1f   = (float*)take((size_t)NG * NFF * 4);

    auto grid1 = [](int n) { return dim3((unsigned)((n + 255) / 256)); };
    auto castb = [&](const float* in, __bf16* out, int cnt_) {
        int c4 = cnt_ / 4;
        cast4_kernel<<<grid1(c4), 256, 0, stream>>>((const float4*)in, (uint2*)out, c4);
    };
    auto gemm = [&](const __bf16* A, const u32* Bp, const float* bias, const float* resid,
                    float* oF, __bf16* oB, int M, int K, int Nc, int act) {
        dim3 g((unsigned)((M + 127) / 128), (unsigned)(Nc / 32));
        dim3 b(256);
#define GEMM_CASE(KK, AA)                                                              \
        if (K == KK && act == AA) {                                                    \
            wmma_gemm_kernel<KK, AA><<<g, b, 0, stream>>>(A, Bp, bias, resid, oF, oB, M, Nc); \
            return;                                                                    \
        }
        GEMM_CASE(128, 0) GEMM_CASE(128, 1) GEMM_CASE(128, 2) GEMM_CASE(128, 4)
        GEMM_CASE(256, 0) GEMM_CASE(256, 1) GEMM_CASE(256, 3)
#undef GEMM_CASE
    };

    // ---- weight pack (fp32 [out,in] -> fragment-ordered bf16 pairs) ----
    for (int l = 0; l < LAYERS; ++l) {
        wpack_kernel<<<grid1(HC * NF / 2), 256, 0, stream>>>(fw1 + (size_t)l * NF * HC, WBf1 + (size_t)l * HC * NF / 2, HC, NF);
        wpack_kernel<<<grid1(NF * NF / 2), 256, 0, stream>>>(fw2 + (size_t)l * NF * NF, WBf2 + (size_t)l * NF * NF / 2, NF, NF);
        wpack_kernel<<<grid1(HC * NF / 2), 256, 0, stream>>>(l1w + (size_t)l * NF * HC, WBl1 + (size_t)l * HC * NF / 2, HC, NF);
        wpack_kernel<<<grid1(NF * HC / 2), 256, 0, stream>>>(l2w + (size_t)l * HC * NF, WBl2 + (size_t)l * NF * HC / 2, NF, HC);
        wpack_kernel<<<grid1(HC * HC / 2), 256, 0, stream>>>(l3w + (size_t)l * HC * HC, WBl3 + (size_t)l * HC * HC / 2, HC, HC);
        wpack_kernel<<<grid1(2 * HC * HC / 2), 256, 0, stream>>>(emw + (size_t)l * HC * 2 * HC, WBem + (size_t)l * 2 * HC * HC / 2, 2 * HC, HC);
    }
    wpack_kernel<<<grid1(HC * NFF / 2), 256, 0, stream>>>(hw1, WBh1, HC, NFF);

    // ---- embedding + edge projection (emit f32 + bf16 in one pass) ----
    embed_kernel<<<grid1(NN * HC), 256, 0, stream>>>(x, pos, vert, posw, hbuf, h_b);
    edgeproj_kernel<<<grid1(NE * HC), 256, 0, stream>>>(eattr, edgew, ea, ea_b);

    // ---- interaction layers ----
    for (int l = 0; l < LAYERS; ++l) {
        // filter MLP:  t1 = ssp(ea @ fw1.T + b1)   [E,256] (bf16)
        gemm(ea_b, WBf1 + (size_t)l * HC * NF / 2, fb1 + (size_t)l * NF, nullptr,
             nullptr, t1_b, NE, HC, NF, 1);
        // W = t1 @ fw2.T + b2   [E,256] (f32)
        gemm(t1_b, WBf2 + (size_t)l * NF * NF / 2, fb2 + (size_t)l * NF, nullptr,
             Wf, nullptr, NE, NF, NF, 0);
        // lin1h = h @ lin1.T   [N,256] (f32, no bias)
        gemm(h_b, WBl1 + (size_t)l * HC * NF / 2, nullptr, nullptr,
             lin1h, nullptr, NN, HC, NF, 0);
        // agg[dst] += lin1h[src] * W
        zero_kernel<<<grid1(NN * NF), 256, 0, stream>>>(agg, NN * NF);
        cfconv_scatter_kernel<<<grid1(NE * NF), 256, 0, stream>>>(lin1h, Wf, src, dst, agg);
        // t2 = ssp(agg @ lin2.T + b)   [N,128] (bf16)
        castb(agg, agg_b, NN * NF);
        gemm(agg_b, WBl2 + (size_t)l * NF * HC / 2, l2b + (size_t)l * HC, nullptr,
             nullptr, t2_b, NN, NF, HC, 1);
        // h = relu(t2 @ lin3.T + b) + h   (fused residual; also emits bf16 h for next layer)
        gemm(t2_b, WBl3 + (size_t)l * HC * HC / 2, l3b + (size_t)l * HC, hbuf,
             hbuf, h_b, NN, HC, HC, 2);
        // edge MLP: ea = tanh([ea, h_src+h_dst] @ em.T + b) + ea  (also emits bf16 ea)
        econcat_kernel<<<grid1(NE * 2 * HC), 256, 0, stream>>>(ea, hbuf, src, dst, Xb);
        gemm(Xb, WBem + (size_t)l * 2 * HC * HC / 2, emb + (size_t)l * HC, ea,
             ea, ea_b, NE, 2 * HC, HC, 3);
    }

    // ---- global mean pool ----
    zero_kernel<<<grid1(NG * HC), 256, 0, stream>>>(ssum, NG * HC);
    zero_kernel<<<grid1(NG), 256, 0, stream>>>(cnt, NG);
    pool_scatter_kernel<<<grid1(NN * HC), 256, 0, stream>>>(hbuf, batch, ssum, cnt);
    pool_fin_kernel<<<grid1(NG * HC), 256, 0, stream>>>(ssum, cnt, g_b);

    // ---- head: Linear(128->512) + exact GELU (WMMA), then Linear(512->1) ----
    gemm(g_b, WBh1, hb1, nullptr, h1f, nullptr, NG, HC, NFF, 4);
    head2_kernel<<<grid1(NG), 256, 0, stream>>>(h1f, hw2, hb2, y);
}
